// HyperSAGNN_5119601017143
// MI455X (gfx1250) — matrix operand order repaired and checked by reading, not compiled, not run on hardware
//
#include <hip/hip_runtime.h>
#include <hip/hip_bf16.h>
#include <math.h>
#include <stdint.h>

// ---------------- problem constants ----------------
#define N_NODES   4096
#define HID       512
#define NUM_HEADS 8
#define HEAD_DIM  64
#define NUM_SETS  1365

// ---------------- CDNA5 vector types ----------------
typedef __attribute__((ext_vector_type(16))) __bf16 v16bf;
typedef __attribute__((ext_vector_type(8)))  __bf16 v8bf;
typedef __attribute__((ext_vector_type(4)))  __bf16 v4bf;
typedef __attribute__((ext_vector_type(2)))  __bf16 v2bf;
typedef __attribute__((ext_vector_type(8)))  float  v8f;
typedef __attribute__((ext_vector_type(4)))  unsigned int u32x4;
typedef __attribute__((ext_vector_type(8)))  int i32x8;
typedef __attribute__((ext_vector_type(4)))  int i32x4;

// ---------------- GEMM tiling ----------------
#define BM 64
#define BN 128
#define BK 32
#define APAD 8   // LDS row = 32 bf16 data (64 B = 16 dw) + 8 bf16 pad (16 B = 4 dw)
#define BPAD 8

enum { EPI_NONE = 0, EPI_RELU = 1, EPI_RESID = 2 };

__device__ __forceinline__ __bf16 f2bf(float f) { return (__bf16)f; }

// ---------------- Tensor Data Mover (gfx1250) ----------------
#if __has_builtin(__builtin_amdgcn_tensor_load_to_lds) && __has_builtin(__builtin_amdgcn_s_wait_tensorcnt)
#define HAVE_TDM 1
#else
#define HAVE_TDM 0
#endif

#if HAVE_TDM
// 2D bf16 tile load: tile_w contiguous elems x tile_h rows, row stride in elems.
// LDS padding: 16 dwords (64B) data -> +4 dwords (16B) pad, matching the
// (BK+PAD) = 40-bf16 LDS row stride of As/BsT.
__device__ __forceinline__ void tdm_load_2d(const void* gaddr, unsigned int lds_off,
                                            unsigned int tile_w, unsigned int tile_h,
                                            unsigned int tensor_w, unsigned int tensor_h,
                                            unsigned int row_stride)
{
    const unsigned long long ga = (unsigned long long)(uintptr_t)gaddr;
    u32x4 g0;
    g0[0] = 1u;                                        // count=1, user mode
    g0[1] = lds_off;                                   // lds_addr (bytes)
    g0[2] = (unsigned int)(ga & 0xffffffffu);          // global_addr[31:0]
    g0[3] = (unsigned int)((ga >> 32) & 0x01ffffffu)   // global_addr[56:32]
          | (2u << 30);                                // type = 2 ("image")
    i32x8 g1;
    g1[0] = (int)((1u << 16)      // data_size = 1 -> 2 bytes
                | (1u << 20)      // pad_enable
                | (3u << 22)      // pad_interval: 16 dwords
                | (3u << 25));    // pad_amount:   4 dwords
    g1[1] = (int)((tensor_w & 0xffffu) << 16);                                   // dim0[15:0]
    g1[2] = (int)(((tensor_w >> 16) & 0xffffu) | ((tensor_h & 0xffffu) << 16));  // dim0 hi | dim1 lo
    g1[3] = (int)(((tensor_h >> 16) & 0xffffu) | ((tile_w & 0xffffu) << 16));    // dim1 hi | tile_dim0
    g1[4] = (int)(tile_h & 0xffffu);                                             // tile_dim1 (tile_dim2=0)
    g1[5] = (int)row_stride;                                                     // dim0_stride[31:0]
    g1[6] = 0;
    g1[7] = 0;
    const i32x4 z4 = {0, 0, 0, 0};
#if defined(__clang_major__) && (__clang_major__ >= 23)
    const i32x8 z8 = {0, 0, 0, 0, 0, 0, 0, 0};
    __builtin_amdgcn_tensor_load_to_lds(g0, g1, z4, z4, z8, 0);
#else
    __builtin_amdgcn_tensor_load_to_lds(g0, g1, z4, z4, 0);
#endif
}
#endif // HAVE_TDM

// ---------------- WMMA fragment loads from padded LDS ----------------
// A fragment 16x32: lane L -> row rowBase+(L&15); lanes 0-15: K 0..7 & 16..23,
// lanes 16-31: K 8..15 & 24..31. Contiguous halves -> ds_load_b128.
__device__ __forceinline__ v16bf load_afrag(const __bf16 As[BM][BK + APAD],
                                            int rowBase, int lane)
{
    const int row = rowBase + (lane & 15);
    const int kh  = (lane >> 4) * 8;
    const v8bf lo = *reinterpret_cast<const v8bf*>(&As[row][kh]);
    const v8bf hi = *reinterpret_cast<const v8bf*>(&As[row][16 + kh]);
    v16bf a;
    #pragma unroll
    for (int e = 0; e < 8; ++e) { a[e] = lo[e]; a[e + 8] = hi[e]; }
    return a;
}

// B fragment 32x16 (K x N) from B-transposed LDS (n, k): lanes 0-15: K 0..7 & 16..23,
// lanes 16-31: K 8..15 & 24..31 (same K order as A).
__device__ __forceinline__ v16bf load_bfrag(const __bf16 BsT[BN][BK + BPAD],
                                            int colBase, int lane)
{
    const int n  = colBase + (lane & 15);
    const int kb = (lane >> 4) * 8;
    const v8bf lo = *reinterpret_cast<const v8bf*>(&BsT[n][kb]);
    const v8bf hi = *reinterpret_cast<const v8bf*>(&BsT[n][16 + kb]);
    v16bf b;
    #pragma unroll
    for (int e = 0; e < 8; ++e) { b[e] = lo[e]; b[e + 8] = hi[e]; }
    return b;
}

// ---------------- GEMM: C = epilogue(Abf @ BT^T + bias) ----------------
// Abf: M x K bf16 row-major (activations), BT: Nn x K bf16 (weights transposed).
// EPI_RESID: C = beta*(acc+bias) + resid.  MIRROR: also emit bf16 copy of C.
template <int MODE, bool MIRROR>
__global__ __launch_bounds__(256)
void gemm_bf16_wmma(const __bf16* __restrict__ Abf,
                    const __bf16* __restrict__ BT,
                    const float* __restrict__ bias,
                    const float* __restrict__ resid,
                    const float* __restrict__ betap,
                    float* __restrict__ C,
                    __bf16* __restrict__ Cbf,
                    int M, int K, int Nn)
{
    __shared__ __align__(16) __bf16 As[2][BM][BK + APAD];
    __shared__ __align__(16) __bf16 BsT[2][BN][BK + BPAD];

    const int tid  = threadIdx.x;
    const int lane = tid & 31;
    const int wave = tid >> 5;
    const int wrow = (wave & 1) * 32;       // 2x4 wave grid over 64x128 C tile
    const int wcol = (wave >> 1) * 32;

    const int m0 = blockIdx.x * BM;
    const int n0 = blockIdx.y * BN;
    const int nT = K / BK;

    v8f c00 = {}, c01 = {}, c10 = {}, c11 = {};

    const __bf16* agbase = Abf + (size_t)m0 * K;
    const __bf16* bgbase = BT + (size_t)n0 * K;

#if HAVE_TDM
    // ---- prologue: TDM tile 0 into buffer 0 ----
    if (wave == 0) {
        tdm_load_2d(agbase, (unsigned int)(uintptr_t)&As[0][0][0],  BK, BM, K, M,  K);
        tdm_load_2d(bgbase, (unsigned int)(uintptr_t)&BsT[0][0][0], BK, BN, K, Nn, K);
        __builtin_amdgcn_s_wait_tensorcnt(0);
    }
    __syncthreads();
    int cur = 0;
    for (int t = 0; t < nT; ++t) {
        const bool more = (t + 1 < nT);
        if (more && wave == 0) {   // async DMA of next tile overlaps WMMA
            tdm_load_2d(agbase + (size_t)(t + 1) * BK,
                        (unsigned int)(uintptr_t)&As[cur ^ 1][0][0], BK, BM, K, M, K);
            tdm_load_2d(bgbase + (size_t)(t + 1) * BK,
                        (unsigned int)(uintptr_t)&BsT[cur ^ 1][0][0], BK, BN, K, Nn, K);
        }
        const v16bf a0 = load_afrag(As[cur], wrow, lane);
        const v16bf a1 = load_afrag(As[cur], wrow + 16, lane);
        const v16bf b0 = load_bfrag(BsT[cur], wcol, lane);
        const v16bf b1 = load_bfrag(BsT[cur], wcol + 16, lane);
        c00 = __builtin_amdgcn_wmma_f32_16x16x32_bf16(false, a0, false, b0, (short)0, c00, false, false);
        c01 = __builtin_amdgcn_wmma_f32_16x16x32_bf16(false, a0, false, b1, (short)0, c01, false, false);
        c10 = __builtin_amdgcn_wmma_f32_16x16x32_bf16(false, a1, false, b0, (short)0, c10, false, false);
        c11 = __builtin_amdgcn_wmma_f32_16x16x32_bf16(false, a1, false, b1, (short)0, c11, false, false);
        if (more && wave == 0) __builtin_amdgcn_s_wait_tensorcnt(0);
        __syncthreads();
        cur ^= 1;
    }
#else
    // ---- manual double-buffered staging (pure bf16 copies, b128 both sides) ----
    const int arow = tid >> 2, akseg = (tid & 3) * 8;   // 8 bf16 per thread (A)
    const int bn = tid & 127, bkh = (tid >> 7) * 16;    // 16 bf16 per thread (B)
    const __bf16* aptr = agbase + (size_t)arow * K + akseg;
    const __bf16* bptr = bgbase + (size_t)bn * K + bkh;
    {
        const v8bf a  = *reinterpret_cast<const v8bf*>(aptr);
        const v8bf p0 = *reinterpret_cast<const v8bf*>(bptr);
        const v8bf p1 = *reinterpret_cast<const v8bf*>(bptr + 8);
        *reinterpret_cast<v8bf*>(&As[0][arow][akseg]) = a;
        *reinterpret_cast<v8bf*>(&BsT[0][bn][bkh])     = p0;
        *reinterpret_cast<v8bf*>(&BsT[0][bn][bkh + 8]) = p1;
    }
    __syncthreads();
    int cur = 0;
    for (int t = 0; t < nT; ++t) {
        const bool more = (t + 1 < nT);
        v8bf na = {}, nb0 = {}, nb1 = {};
        if (more) {
            na  = *reinterpret_cast<const v8bf*>(aptr + (t + 1) * BK);
            nb0 = *reinterpret_cast<const v8bf*>(bptr + (t + 1) * BK);
            nb1 = *reinterpret_cast<const v8bf*>(bptr + (t + 1) * BK + 8);
        }
        const v16bf a0 = load_afrag(As[cur], wrow, lane);
        const v16bf a1 = load_afrag(As[cur], wrow + 16, lane);
        const v16bf b0 = load_bfrag(BsT[cur], wcol, lane);
        const v16bf b1 = load_bfrag(BsT[cur], wcol + 16, lane);
        c00 = __builtin_amdgcn_wmma_f32_16x16x32_bf16(false, a0, false, b0, (short)0, c00, false, false);
        c01 = __builtin_amdgcn_wmma_f32_16x16x32_bf16(false, a0, false, b1, (short)0, c01, false, false);
        c10 = __builtin_amdgcn_wmma_f32_16x16x32_bf16(false, a1, false, b0, (short)0, c10, false, false);
        c11 = __builtin_amdgcn_wmma_f32_16x16x32_bf16(false, a1, false, b1, (short)0, c11, false, false);
        if (more) {
            *reinterpret_cast<v8bf*>(&As[cur ^ 1][arow][akseg]) = na;
            *reinterpret_cast<v8bf*>(&BsT[cur ^ 1][bn][bkh])     = nb0;
            *reinterpret_cast<v8bf*>(&BsT[cur ^ 1][bn][bkh + 8]) = nb1;
        }
        __syncthreads();
        cur ^= 1;
    }
#endif

    // ---- epilogue; C 16x16 f32 layout: VGPR e -> M = e + (lane<16 ? 0 : 8) ----
    const float beta = (MODE == EPI_RESID) ? betap[0] : 1.0f;
    const int rb = m0 + wrow + (lane >> 4) * 8;
    const int cb = n0 + wcol + (lane & 15);
    const v8f* cc[2][2] = {{&c00, &c01}, {&c10, &c11}};
    #pragma unroll
    for (int r = 0; r < 2; ++r) {
        #pragma unroll
        for (int s = 0; s < 2; ++s) {
            const v8f& c = *cc[r][s];
            const int gc = cb + s * 16;
            const float bv = bias[gc];
            #pragma unroll
            for (int e = 0; e < 8; ++e) {
                const int gr = rb + r * 16 + e;
                float val = c[e] + bv;
                if (MODE == EPI_RELU)  val = fmaxf(val, 0.0f);
                if (MODE == EPI_RESID) val = beta * val + resid[(size_t)gr * Nn + gc];
                C[(size_t)gr * Nn + gc] = val;
                if (MIRROR) Cbf[(size_t)gr * Nn + gc] = f2bf(val);
            }
        }
    }
}

// ---------------- one-time weight transpose+convert: WT[n][k] = bf16(W[k][n]) ----
__global__ __launch_bounds__(256)
void transpose_w_bf16(const float* __restrict__ W, __bf16* __restrict__ WT)
{
    __shared__ float tile[32][33];
    const int tx = threadIdx.x & 31, ty = threadIdx.x >> 5;  // 32 x 8
    const int kb = blockIdx.x * 32, nb = blockIdx.y * 32;
    #pragma unroll
    for (int j = 0; j < 32; j += 8)
        tile[ty + j][tx] = W[(size_t)(kb + ty + j) * HID + nb + tx];
    __syncthreads();
    #pragma unroll
    for (int j = 0; j < 32; j += 8)
        WT[(size_t)(nb + ty + j) * HID + kb + tx] = f2bf(tile[tx][ty + j]);
}

// ---------------- fp32 -> bf16 row-mirror convert ----------------
__global__ __launch_bounds__(256)
void cvt_f32_to_bf16(const float* __restrict__ x, __bf16* __restrict__ y)
{
    const int i = blockIdx.x * 256 + threadIdx.x;
    const float4 f = reinterpret_cast<const float4*>(x)[i];
    v4bf p; p[0] = f2bf(f.x); p[1] = f2bf(f.y); p[2] = f2bf(f.z); p[3] = f2bf(f.w);
    reinterpret_cast<v4bf*>(y)[i] = p;
}

// ---------------- masked attention over tiny contiguous segments ----------------
// batch_indices sorted -> contiguous sets (avg ~3 nodes). 1 wave per (node, head),
// online softmax; also emits bf16 mirror for the following Wo GEMM.
__global__ __launch_bounds__(256)
void attn_small_seg(const float* __restrict__ q,
                    const float* __restrict__ k,
                    const float* __restrict__ v,
                    const int* __restrict__ bi,
                    float* __restrict__ out,
                    __bf16* __restrict__ outbf, int n)
{
    const int i    = blockIdx.x;
    const int h    = threadIdx.x >> 5;
    const int lane = threadIdx.x & 31;
    const int base = i * HID + h * HEAD_DIM;

    const float2 qv = *reinterpret_cast<const float2*>(&q[base + 2 * lane]);
    const int myset = bi[i];

    int lo = 0, hb = n;
    while (lo < hb) { int mid = (lo + hb) >> 1; if (bi[mid] < myset) lo = mid + 1; else hb = mid; }
    const int start = lo;
    lo = 0; hb = n;
    while (lo < hb) { int mid = (lo + hb) >> 1; if (bi[mid] <= myset) lo = mid + 1; else hb = mid; }
    const int end = lo;

    const float scale = 0.125f;  // 1/sqrt(HEAD_DIM)
    float m = -__builtin_inff(), l = 0.f, a0 = 0.f, a1 = 0.f;

    for (int j = start; j < end; ++j) {
        if (j == i) continue;
        const int jb = j * HID + h * HEAD_DIM;
        const float2 kv = *reinterpret_cast<const float2*>(&k[jb + 2 * lane]);
        const float2 vv = *reinterpret_cast<const float2*>(&v[jb + 2 * lane]);
        float p = qv.x * kv.x + qv.y * kv.y;
        #pragma unroll
        for (int off = 16; off >= 1; off >>= 1) p += __shfl_xor(p, off, 32);
        const float s  = p * scale;
        const float mn = fmaxf(m, s);
        const float corr = __expf(m - mn);
        const float w    = __expf(s - mn);
        l  = l  * corr + w;
        a0 = a0 * corr + w * vv.x;
        a1 = a1 * corr + w * vv.y;
        m = mn;
    }
    const float inv = (l > 0.f) ? 1.0f / l : 0.0f;  // empty row -> 0 (nan_to_num)
    float2 o; o.x = a0 * inv; o.y = a1 * inv;
    *reinterpret_cast<float2*>(&out[base + 2 * lane]) = o;
    v2bf ob; ob[0] = f2bf(o.x); ob[1] = f2bf(o.y);
    *reinterpret_cast<v2bf*>(&outbf[base + 2 * lane]) = ob;
}

// ---------------- per-set mean of squared difference ----------------
__global__ __launch_bounds__(256)
void segment_mean_sq(const float* __restrict__ dyn,
                     const float* __restrict__ se,
                     const int* __restrict__ bi,
                     float* __restrict__ out, int n)
{
    const int s = blockIdx.x;
    int lo = 0, hb = n;
    while (lo < hb) { int mid = (lo + hb) >> 1; if (bi[mid] < s) lo = mid + 1; else hb = mid; }
    const int start = lo;
    lo = 0; hb = n;
    while (lo < hb) { int mid = (lo + hb) >> 1; if (bi[mid] <= s) lo = mid + 1; else hb = mid; }
    const int end = lo;

    const float cnt = fmaxf((float)(end - start), 1.0f);
    const int d = threadIdx.x * 2;
    float acc0 = 0.f, acc1 = 0.f;
    for (int mm = start; mm < end; ++mm) {
        const float2 dv = *reinterpret_cast<const float2*>(&dyn[(size_t)mm * HID + d]);
        const float2 sv = *reinterpret_cast<const float2*>(&se[(size_t)mm * HID + d]);
        const float e0 = dv.x - sv.x, e1 = dv.y - sv.y;
        acc0 += e0 * e0; acc1 += e1 * e1;
    }
    float2 o; o.x = acc0 / cnt; o.y = acc1 / cnt;
    *reinterpret_cast<float2*>(&out[(size_t)s * HID + d]) = o;
}

// ---------------- host driver ----------------
extern "C" void kernel_launch(void* const* d_in, const int* in_sizes, int n_in,
                              void* d_out, int out_size, void* d_ws, size_t ws_size,
                              hipStream_t stream)
{
    const float* emb  = (const float*)d_in[0];
    const int*   bi   = (const int*)  d_in[1];
    const float* W[9] = { (const float*)d_in[2],                      // Ws
                          (const float*)d_in[4],  (const float*)d_in[6],
                          (const float*)d_in[8],  (const float*)d_in[10],   // q1,k1,v1,o1
                          (const float*)d_in[13], (const float*)d_in[15],
                          (const float*)d_in[17], (const float*)d_in[19] }; // q2,k2,v2,o2
    const float* bs   = (const float*)d_in[3];
    const float* bq1  = (const float*)d_in[5];   const float* bk1 = (const float*)d_in[7];
    const float* bv1  = (const float*)d_in[9];   const float* bo1 = (const float*)d_in[11];
    const float* be1  = (const float*)d_in[12];
    const float* bq2  = (const float*)d_in[14];  const float* bk2 = (const float*)d_in[16];
    const float* bv2  = (const float*)d_in[18];  const float* bo2 = (const float*)d_in[20];
    const float* be2  = (const float*)d_in[21];

    const size_t NH = (size_t)N_NODES * HID;
    const size_t WE = (size_t)HID * HID;
    float* wsf = (float*)d_ws;
    float* se = wsf;            // fp32 region: 7 * NH floats
    float* qb = wsf + 1 * NH;
    float* kb = wsf + 2 * NH;
    float* vb = wsf + 3 * NH;
    float* ao = wsf + 4 * NH;
    float* d1 = wsf + 5 * NH;
    float* d2 = wsf + 6 * NH;
    __bf16* bfr   = (__bf16*)(wsf + 7 * NH);   // bf16 region
    __bf16* embbf = bfr;
    __bf16* aobf  = bfr + 1 * NH;
    __bf16* d1bf  = bfr + 2 * NH;
    __bf16* wt    = bfr + 3 * NH;              // 9 transposed bf16 weights
    __bf16* WT[9];
    for (int i = 0; i < 9; ++i) WT[i] = wt + (size_t)i * WE;

    // one-time converts
    for (int i = 0; i < 9; ++i)
        transpose_w_bf16<<<dim3(HID / 32, HID / 32), 256, 0, stream>>>(W[i], WT[i]);
    cvt_f32_to_bf16<<<NH / 4 / 256, 256, 0, stream>>>(emb, embbf);

    const dim3 gg(N_NODES / BM, HID / BN);
    const dim3 bb(256);

    // static branch: se = relu(emb @ Ws + bs)
    gemm_bf16_wmma<EPI_RELU, false><<<gg, bb, 0, stream>>>(embbf, WT[0], bs, nullptr, nullptr, se, nullptr, N_NODES, HID, HID);

    // ---- attention layer 1 (x = emb) ----
    gemm_bf16_wmma<EPI_NONE, false><<<gg, bb, 0, stream>>>(embbf, WT[1], bq1, nullptr, nullptr, qb, nullptr, N_NODES, HID, HID);
    gemm_bf16_wmma<EPI_NONE, false><<<gg, bb, 0, stream>>>(embbf, WT[2], bk1, nullptr, nullptr, kb, nullptr, N_NODES, HID, HID);
    gemm_bf16_wmma<EPI_NONE, false><<<gg, bb, 0, stream>>>(embbf, WT[3], bv1, nullptr, nullptr, vb, nullptr, N_NODES, HID, HID);
    attn_small_seg<<<N_NODES, 256, 0, stream>>>(qb, kb, vb, bi, ao, aobf, N_NODES);
    gemm_bf16_wmma<EPI_RESID, true><<<gg, bb, 0, stream>>>(aobf, WT[4], bo1, emb, be1, d1, d1bf, N_NODES, HID, HID);

    // ---- attention layer 2 (x = d1) ----
    gemm_bf16_wmma<EPI_NONE, false><<<gg, bb, 0, stream>>>(d1bf, WT[5], bq2, nullptr, nullptr, qb, nullptr, N_NODES, HID, HID);
    gemm_bf16_wmma<EPI_NONE, false><<<gg, bb, 0, stream>>>(d1bf, WT[6], bk2, nullptr, nullptr, kb, nullptr, N_NODES, HID, HID);
    gemm_bf16_wmma<EPI_NONE, false><<<gg, bb, 0, stream>>>(d1bf, WT[7], bv2, nullptr, nullptr, vb, nullptr, N_NODES, HID, HID);
    attn_small_seg<<<N_NODES, 256, 0, stream>>>(qb, kb, vb, bi, ao, aobf, N_NODES);
    gemm_bf16_wmma<EPI_RESID, false><<<gg, bb, 0, stream>>>(aobf, WT[8], bo2, d1, be2, d2, nullptr, N_NODES, HID, HID);

    // ---- output: per-set mean squared difference ----
    segment_mean_sq<<<NUM_SETS, 256, 0, stream>>>(d2, se, bi, (float*)d_out, N_NODES);
}
